// ConflateLoss_45380624449754
// MI455X (gfx1250) — matrix operand order repaired and checked by reading, not compiled, not run on hardware
//
#include <hip/hip_runtime.h>
#include <hip/hip_bf16.h>
#include <math.h>

typedef __attribute__((ext_vector_type(16))) _Float16 v16h;
typedef __attribute__((ext_vector_type(8)))  float    v8f;
typedef __attribute__((ext_vector_type(2)))  float    vf2;

#define BROWS      16384
#define DIM        128
#define NNEG       49
#define NCAND      50
#define GAMMA_F    10.0f
#define EPS_F      1e-8f
#define INV_LN10   0.43429448190325176f
#define WPB        4          // waves per block (wave32)

// One wave per 16-anchor tile. Dot products via v_wmma_f32_16x16x32_f16
// (diagonal of A[16xK] x B[Kx16] where column n = candidate j of anchor n).
// Norms in fp32, online softmax in fp32, one partial per tile -> d_ws.
__global__ __launch_bounds__(WPB * 32)
void conflate_main(const float* __restrict__ anchor,
                   const float* __restrict__ match,
                   const int*   __restrict__ negidx,
                   float*       __restrict__ partials)
{
    const int lane = threadIdx.x & 31;
    const int wave = threadIdx.x >> 5;
    const int tile = blockIdx.x * WPB + wave;
    const int base = tile * 16;
    const bool hi  = lane >= 16;
    const int  col = lane & 15;       // A-row / B-column owned by this lane pair
    const int  row = base + col;      // global anchor row for this column

    // ---- Build A fragments (anchor tile, f32 -> f16) + anchor sumsq ----
    v16h afrag[4];
    float asq = 0.0f;
    const float* arow = anchor + (size_t)row * DIM;
    #pragma unroll
    for (int c = 0; c < 4; ++c) {
        #pragma unroll
        for (int d = 0; d < 8; ++d) {
            const int k = 32 * c + ((d & 4) ? 16 : 0) + 2 * (d & 3) + (hi ? 8 : 0);
            vf2 v = *(const vf2*)(arow + k);   // 8B-aligned: k even, row 512B-aligned
            asq = fmaf(v.x, v.x, asq);
            asq = fmaf(v.y, v.y, asq);
            afrag[c][2 * d]     = (_Float16)v.x;
            afrag[c][2 * d + 1] = (_Float16)v.y;
        }
    }
    asq += __shfl_xor(asq, 16);               // lanes l and l+16 cover full row
    const float a_norm = sqrtf(asq);

    // ---- Online softmax state (per diagonal lane) ----
    float m = 0.0f, s = 1.0f, l0 = 0.0f;

    for (int j = 0; j < NCAND; ++j) {
        const int ci = (j == 0) ? row
                                : negidx[(size_t)row * NNEG + (j - 1)];
        const float* crow = match + (size_t)ci * DIM;

        float csq = 0.0f;
        v8f acc = {};
        #pragma unroll
        for (int c = 0; c < 4; ++c) {
            v16h bfrag;
            #pragma unroll
            for (int d = 0; d < 8; ++d) {
                const int k = 32 * c + ((d & 4) ? 16 : 0) + 2 * (d & 3) + (hi ? 8 : 0);
                vf2 v = *(const vf2*)(crow + k);
                csq = fmaf(v.x, v.x, csq);
                csq = fmaf(v.y, v.y, csq);
                bfrag[2 * d]     = (_Float16)v.x;
                bfrag[2 * d + 1] = (_Float16)v.y;
            }
            acc = __builtin_amdgcn_wmma_f32_16x16x32_f16(
                      /*neg_a=*/false, afrag[c],
                      /*neg_b=*/false, bfrag,
                      /*c_mod=*/(short)0, acc,
                      /*reuse_a=*/false, /*reuse_b=*/false);
        }
        csq += __shfl_xor(csq, 16);           // full candidate-row sumsq

        // Diagonal extraction: rows 0-7 in lanes 0-7 (VGPR=lane),
        // rows 8-15 in lanes 24-31 (VGPR=lane-24). r = lane & 7 either way.
        const int r = lane & 7;
        float dot = acc[0];
        #pragma unroll
        for (int i = 1; i < 8; ++i)
            if (r == i) dot = acc[i];

        const float sim = dot / fmaxf(a_norm * sqrtf(csq), EPS_F);
        const float l   = GAMMA_F * sim;

        if (j == 0) {
            l0 = l; m = l; s = 1.0f;
        } else {
            const float nm = fmaxf(m, l);
            s = s * __expf(m - nm) + __expf(l - nm);
            m = nm;
        }
    }

    const bool valid = (lane < 8) || (lane >= 24);
    float contrib = valid ? (-(l0 - m - __logf(s)) * INV_LN10) : 0.0f;

    // Wave-wide sum (16 valid contributions)
    #pragma unroll
    for (int off = 16; off >= 1; off >>= 1)
        contrib += __shfl_xor(contrib, off);

    if (lane == 0)
        partials[tile] = contrib;
}

// Deterministic single-block reduction of the per-tile partials.
__global__ __launch_bounds__(256)
void conflate_reduce(const float* __restrict__ partials,
                     float* __restrict__ out, int n)
{
    __shared__ float sm[256];
    float v = 0.0f;
    for (int i = threadIdx.x; i < n; i += 256)
        v += partials[i];
    sm[threadIdx.x] = v;
    __syncthreads();
    #pragma unroll
    for (int s = 128; s > 0; s >>= 1) {
        if ((int)threadIdx.x < s) sm[threadIdx.x] += sm[threadIdx.x + s];
        __syncthreads();
    }
    if (threadIdx.x == 0)
        out[0] = sm[0];
}

extern "C" void kernel_launch(void* const* d_in, const int* in_sizes, int n_in,
                              void* d_out, int out_size, void* d_ws, size_t ws_size,
                              hipStream_t stream)
{
    const float* anchor = (const float*)d_in[0];
    const float* match  = (const float*)d_in[1];
    const int*   negidx = (const int*)d_in[2];
    float* partials = (float*)d_ws;           // tiles * 4 bytes = 4 KB
    float* out      = (float*)d_out;

    const int tiles = BROWS / 16;             // 1024 waves
    conflate_main<<<dim3(tiles / WPB), dim3(WPB * 32), 0, stream>>>(
        anchor, match, negidx, partials);
    conflate_reduce<<<dim3(1), dim3(256), 0, stream>>>(partials, out, tiles);
}
